// FourierTransformerLayer_71305047048418
// MI455X (gfx1250) — matrix operand order repaired and checked by reading, not compiled, not run on hardware
//
#include <hip/hip_runtime.h>
#include <cstdint>
#include <cstddef>

// ---------------------------------------------------------------- constants
constexpr int kB = 4;
constexpr int kS = 2048;
constexpr int kD = 512;
constexpr int kBS = kB * kS;          // 8192 rows
constexpr size_t kBSD = (size_t)kB * kS * kD;

typedef __bf16 bf16_t;
typedef __attribute__((ext_vector_type(16))) __bf16 v16bf;
typedef __attribute__((ext_vector_type(8)))  __bf16 v8bf;
typedef __attribute__((ext_vector_type(8)))  float  v8f;

// ---------------------------------------------------------------- WMMA utils
__device__ __forceinline__ v8f wmma_bf16(v16bf a, v16bf b, v8f c) {
  // D = A(16x32 bf16) * B(32x16 bf16) + C(16x16 f32)
  return __builtin_amdgcn_wmma_f32_16x16x32_bf16(false, a, false, b,
                                                 (short)0, c, false, false);
}

// A fragment (16x32 bf16). Element (m,k) at base[m*ldk + k].
// wave32 layout: lanes 0-15 -> rows; elems 0..15 = K b0..b0+15
// with b0 = (lane<16) ? 0 : 16.
__device__ __forceinline__ v16bf wmma_load_a(const bf16_t* base, int ldk, int lane) {
  const int m  = lane & 15;
  const int b0 = (lane < 16) ? 0 : 16;
  const bf16_t* p = base + (size_t)m * ldk;
  v16bf a;
#pragma unroll
  for (int j = 0; j < 16; ++j) a[j] = p[b0 + j];
  return a;
}

// B fragment (32x16 bf16). Element (k,n) at base[n*ldk + k] (K contiguous).
// lanes 0-15: K=0..15, lanes 16-31: K=16..31; col N = lane&15.
__device__ __forceinline__ v16bf wmma_load_b(const bf16_t* base, int ldk, int lane) {
  const int n   = lane & 15;
  const int klo = (lane < 16) ? 0 : 16;
  const bf16_t* p = base + (size_t)n * ldk + klo;
  v16bf b;
#pragma unroll
  for (int j = 0; j < 16; ++j) b[j] = p[j];
  return b;
}

// ---------------------------------------------------------------- FFT utils
__device__ __forceinline__ float2 cmulf(float2 a, float2 b) {
  return make_float2(a.x * b.x - a.y * b.y, a.x * b.y + a.y * b.x);
}

// In-place radix-2 shared-memory FFT. sign=-1 forward, +1 inverse (unscaled).
// Caller must __syncthreads() after last write to buf before calling.
template <int N, int LOGN>
__device__ void fft_lds(float2* buf, int tid, int nthr, float sign) {
  for (int i = tid; i < N; i += nthr) {
    unsigned j = __brev((unsigned)i) >> (32 - LOGN);
    if (j > (unsigned)i) {
      float2 t = buf[i]; buf[i] = buf[j]; buf[j] = t;
    }
  }
  __syncthreads();
  for (int half = 1; half < N; half <<= 1) {
    const float ang = sign * 3.14159265358979323846f / (float)half;
    for (int bi = tid; bi < (N >> 1); bi += nthr) {
      const int blk = bi / half;
      const int off = bi - blk * half;
      const int i0 = blk * (half << 1) + off;
      const int i1 = i0 + half;
      float sn, cs;
      __sincosf(ang * (float)off, &sn, &cs);
      const float2 w = make_float2(cs, sn);
      const float2 u = buf[i0];
      const float2 t = cmulf(w, buf[i1]);
      buf[i0] = make_float2(u.x + t.x, u.y + t.y);
      buf[i1] = make_float2(u.x - t.x, u.y - t.y);
    }
    __syncthreads();
  }
}

// ---------------------------------------------------------------- kernel 1
// Circulant matrices: h = Re(ifft(w)); Mt[n][m] = h[(n-m) mod D] so that
// y[n] = sum_m x[m]*Mt[n][m]; row-major [n][m] -> B fragments contiguous in K.
__global__ void k_build_circulant(
    const float* wr0, const float* wi0, bf16_t* m0,
    const float* wr1, const float* wi1, bf16_t* m1,
    const float* wr2, const float* wi2, bf16_t* m2,
    const float* wr3, const float* wi3, bf16_t* m3) {
  __shared__ float h[kD];
  const float* wr; const float* wi; bf16_t* mt;
  switch (blockIdx.x) {
    case 0:  wr = wr0; wi = wi0; mt = m0; break;
    case 1:  wr = wr1; wi = wi1; mt = m1; break;
    case 2:  wr = wr2; wi = wi2; mt = m2; break;
    default: wr = wr3; wi = wi3; mt = m3; break;
  }
  const int tid = threadIdx.x;  // 0..511
  {
    const float step = 6.28318530717958647692f / (float)kD;
    float acc = 0.f;
    for (int k = 0; k < kD; ++k) {
      const int idx = (k * tid) & (kD - 1);  // arg reduction mod 2*pi
      float sn, cs;
      __sincosf(step * (float)idx, &sn, &cs);
      acc += wr[k] * cs - wi[k] * sn;
    }
    h[tid] = acc * (1.f / (float)kD);
  }
  __syncthreads();
  const int n = tid;
  for (int m = 0; m < kD; ++m)
    mt[(size_t)n * kD + m] = (bf16_t)h[(n - m + kD) & (kD - 1)];
}

// ---------------------------------------------------------------- kernel 2
// xa = DyT_a(x); Xf = fft(xa, axis=D). One block per (b,s) row.
__global__ void k_dyt_fft_d(const float* __restrict__ x,
                            const float* __restrict__ alpha,
                            const float* __restrict__ w,
                            const float* __restrict__ bia,
                            float2* __restrict__ Xf) {
  __shared__ float2 buf[kD];
  const int row = blockIdx.x;
  const int tid = threadIdx.x;
  const float al = alpha[0];
  const float* xr = x + (size_t)row * kD;
  for (int i = tid; i < kD; i += 256) {
    const float v = tanhf(al * xr[i]) * w[i] + bia[i];
    buf[i] = make_float2(v, 0.f);
  }
  __syncthreads();
  fft_lds<kD, 9>(buf, tid, 256, -1.f);
  float2* o = Xf + (size_t)row * kD;
  for (int i = tid; i < kD; i += 256) o[i] = buf[i];
}

// ---------------------------------------------------------------- kernel T
// LDS-tiled complex transpose, per batch: in [R][C] -> out [C][R].
// 32x32 tiles, 256 threads, fully coalesced both sides.
__global__ void k_transpose_c(const float2* __restrict__ in,
                              float2* __restrict__ out, int R, int C) {
  __shared__ float2 tile[32][33];
  const int c0 = blockIdx.x * 32;
  const int r0 = blockIdx.y * 32;
  const size_t bb = (size_t)blockIdx.z * R * C;
  const int tx = threadIdx.x & 31;
  const int ty = threadIdx.x >> 5;  // 0..7
#pragma unroll
  for (int j = 0; j < 32; j += 8)
    tile[ty + j][tx] = in[bb + (size_t)(r0 + ty + j) * C + (c0 + tx)];
  __syncthreads();
#pragma unroll
  for (int j = 0; j < 32; j += 8)
    out[bb + (size_t)(c0 + ty + j) * R + (r0 + tx)] = tile[tx][ty + j];
}

// ---------------------------------------------------------------- kernel 3
// Per (b,d) column of Xt [B][D][S] (contiguous!): q=fft_S(X*qw),
// k=fft_S(X*kw), v=fft_S(X*vw); rt = q/(k+eps)*v; Yt = ifft_S(rt)*rtw.
__global__ void k_attn_a_scol(const float2* __restrict__ Xt,
                              const float* __restrict__ qr, const float* __restrict__ qi,
                              const float* __restrict__ kr, const float* __restrict__ ki,
                              const float* __restrict__ vr, const float* __restrict__ vi,
                              const float* __restrict__ rtr, const float* __restrict__ rti,
                              float2* __restrict__ Yt) {
  __shared__ float2 bx[kS];
  __shared__ float2 ba[kS];
  __shared__ float2 bb[kS];
  const int col = blockIdx.x;          // 0 .. B*D-1
  const int d = col & (kD - 1);
  const int tid = threadIdx.x;         // 512
  const float2 qw = make_float2(qr[d], qi[d]);
  const float2 kw = make_float2(kr[d], ki[d]);
  const float2 vw = make_float2(vr[d], vi[d]);
  const float2 rw = make_float2(rtr[d], rti[d]);
  const float2* xin = Xt + (size_t)col * kS;

  for (int s = tid; s < kS; s += 512) bx[s] = xin[s];
  __syncthreads();
  // k = fft(Xcol * kw)
  for (int s = tid; s < kS; s += 512) ba[s] = cmulf(bx[s], kw);
  __syncthreads();
  fft_lds<kS, 11>(ba, tid, 512, -1.f);
  // q = fft(Xcol * qw)
  for (int s = tid; s < kS; s += 512) bb[s] = cmulf(bx[s], qw);
  __syncthreads();
  fft_lds<kS, 11>(bb, tid, 512, -1.f);
  // bb = q / (k + 1e-12)
  for (int s = tid; s < kS; s += 512) {
    float2 kk = ba[s];
    kk.x += 1e-12f;
    const float den = kk.x * kk.x + kk.y * kk.y;
    const float2 qq = bb[s];
    bb[s] = make_float2((qq.x * kk.x + qq.y * kk.y) / den,
                        (qq.y * kk.x - qq.x * kk.y) / den);
  }
  __syncthreads();
  // v = fft(Xcol * vw); bb *= v
  for (int s = tid; s < kS; s += 512) ba[s] = cmulf(bx[s], vw);
  __syncthreads();
  fft_lds<kS, 11>(ba, tid, 512, -1.f);
  for (int s = tid; s < kS; s += 512) bb[s] = cmulf(bb[s], ba[s]);
  __syncthreads();
  // ifft along S (1/S folded into output), then * rtw
  fft_lds<kS, 11>(bb, tid, 512, +1.f);
  float2* yo = Yt + (size_t)col * kS;
  const float inv = 1.f / (float)kS;
  for (int s = tid; s < kS; s += 512) {
    const float2 r = cmulf(bb[s], rw);
    yo[s] = make_float2(r.x * inv, r.y * inv);
  }
}

// ---------------------------------------------------------------- kernel 4
// x1 = Re(ifft_D(Yf)) + x  ;  xb = bf16(DyT_b(x1)).  Block per row.
__global__ void k_ifft_d_res(const float2* __restrict__ Yf,
                             const float* __restrict__ x,
                             const float* __restrict__ alpha,
                             const float* __restrict__ w,
                             const float* __restrict__ bia,
                             float* __restrict__ x1,
                             bf16_t* __restrict__ xb) {
  __shared__ float2 buf[kD];
  const int row = blockIdx.x;
  const int tid = threadIdx.x;
  const float2* yr = Yf + (size_t)row * kD;
  for (int i = tid; i < kD; i += 256) buf[i] = yr[i];
  __syncthreads();
  fft_lds<kD, 9>(buf, tid, 256, +1.f);
  const float inv = 1.f / (float)kD;
  const float al = alpha[0];
  for (int i = tid; i < kD; i += 256) {
    const size_t g = (size_t)row * kD + i;
    const float v = buf[i].x * inv + x[g];
    x1[g] = v;
    xb[g] = (bf16_t)(tanhf(al * v) * w[i] + bia[i]);
  }
}

// ---------------------------------------------------------------- kernel 5
// Q/K = xb @ circulant(W) row-major; V stored transposed Vt[B][D][S] so the
// flash-attention B fragments of P@V are K-contiguous. 8 waves = 4(M) x 2(N),
// block tile 64x32, per-wave 16x16, K=512 in steps of 32.
__global__ void k_qkv_gemm(const bf16_t* __restrict__ A,
                           const bf16_t* __restrict__ Wq,
                           const bf16_t* __restrict__ Wk,
                           const bf16_t* __restrict__ Wv,
                           bf16_t* __restrict__ Q,
                           bf16_t* __restrict__ K,
                           bf16_t* __restrict__ Vt) {
  const int lane = threadIdx.x & 31;
  const int wave = threadIdx.x >> 5;
  const int row0 = blockIdx.y * 64 + (wave & 3) * 16;
  const int col0 = blockIdx.x * 32 + (wave >> 2) * 16;
  v8f cq = {}, ck = {}, cv = {};
  for (int kk = 0; kk < kD; kk += 32) {
    __builtin_prefetch(A + (size_t)row0 * kD + kk + 64, 0, 1);
    const v16bf a  = wmma_load_a(A + (size_t)row0 * kD + kk, kD, lane);
    const v16bf bq = wmma_load_b(Wq + (size_t)col0 * kD + kk, kD, lane);
    const v16bf bk = wmma_load_b(Wk + (size_t)col0 * kD + kk, kD, lane);
    const v16bf bv = wmma_load_b(Wv + (size_t)col0 * kD + kk, kD, lane);
    cq = wmma_bf16(a, bq, cq);
    ck = wmma_bf16(a, bk, ck);
    cv = wmma_bf16(a, bv, cv);
  }
  const int n = lane & 15;
  const int moff = (lane < 16) ? 0 : 8;
#pragma unroll
  for (int j = 0; j < 8; ++j) {
    const size_t g = (size_t)(row0 + moff + j) * kD + (col0 + n);
    Q[g] = (bf16_t)cq[j];
    K[g] = (bf16_t)ck[j];
  }
  // packed transposed V store: lane's 8 rows are contiguous along S
  {
    const int bidx = row0 / kS;
    const int srow = (row0 & (kS - 1)) + moff;
    v8bf vp;
#pragma unroll
    for (int j = 0; j < 8; ++j) vp[j] = (bf16_t)cv[j];
    *(v8bf*)(Vt + ((size_t)bidx * kD + (col0 + n)) * kS + srow) = vp;
  }
}

// ---------------------------------------------------------------- kernel 6
// Flash attention, block per (b, 16-row q tile). 8 waves split D=512 into
// 64-col chunks; partial scores reduced in LDS; wave 0 runs online softmax;
// epilogue fuses residual (+x1) and DyT_f -> xf (bf16).
__global__ void k_flash_attn(const bf16_t* __restrict__ Qb,
                             const bf16_t* __restrict__ Kb,
                             const bf16_t* __restrict__ Vt,
                             const float* __restrict__ x1,
                             const float* __restrict__ alphaF,
                             const float* __restrict__ wF,
                             const float* __restrict__ bF,
                             float* __restrict__ x2,
                             bf16_t* __restrict__ xfb) {
  __shared__ bf16_t qs[16 * kD];            // 16 KB
  __shared__ float  sred[8][16][32];        // 16 KB
  __shared__ bf16_t Pt[16 * 32];
  __shared__ float  alphas[16], mrow[16], lrow[16];

  const int b = blockIdx.y;
  const int row0 = blockIdx.x * 16;
  const int tid = threadIdx.x;
  const int lane = tid & 31;
  const int wave = tid >> 5;
  const bf16_t* Qp = Qb + (size_t)b * kS * kD;
  const bf16_t* Kp = Kb + (size_t)b * kS * kD;
  const bf16_t* Vp = Vt + (size_t)b * kD * kS;   // transposed [D][S]

  for (int i = tid; i < 16 * kD; i += 256)
    qs[i] = Qp[(size_t)(row0 + (i >> 9)) * kD + (i & (kD - 1))];
  if (tid < 16) { mrow[tid] = -1e30f; lrow[tid] = 0.f; }
  __syncthreads();

  const int dc0 = wave * 64;                // this wave's D chunk
  v8f acc[4] = {v8f{}, v8f{}, v8f{}, v8f{}};
  const float scale = 0.044194173824159216f;  // 1/sqrt(512)

  for (int jt = 0; jt < kS; jt += 32) {
    v8f s0 = {}, s1 = {};
#pragma unroll
    for (int kk = 0; kk < 64; kk += 32) {
      const v16bf a  = wmma_load_a(qs + dc0 + kk, kD, lane);
      const v16bf b0 = wmma_load_b(Kp + (size_t)jt * kD + dc0 + kk, kD, lane);
      const v16bf b1 = wmma_load_b(Kp + (size_t)(jt + 16) * kD + dc0 + kk, kD, lane);
      s0 = wmma_bf16(a, b0, s0);
      s1 = wmma_bf16(a, b1, s1);
    }
    {
      const int n = lane & 15;
      const int moff = (lane < 16) ? 0 : 8;
#pragma unroll
      for (int j = 0; j < 8; ++j) {
        sred[wave][moff + j][n]      = s0[j];
        sred[wave][moff + j][16 + n] = s1[j];
      }
    }
    __syncthreads();

    if (wave == 0 && lane < 16) {            // online softmax, one row/lane
      const int m = lane;
      float srow[32];
      float tmax = -1e30f;
#pragma unroll
      for (int nn = 0; nn < 32; ++nn) {
        float t = 0.f;
#pragma unroll
        for (int w2 = 0; w2 < 8; ++w2) t += sred[w2][m][nn];
        t *= scale;
        srow[nn] = t;
        tmax = fmaxf(tmax, t);
      }
      const float mold = mrow[m];
      const float mnew = fmaxf(mold, tmax);
      const float al = __expf(mold - mnew);
      float ls = 0.f;
#pragma unroll
      for (int nn = 0; nn < 32; ++nn) {
        const float p = __expf(srow[nn] - mnew);
        ls += p;
        Pt[m * 32 + nn] = (bf16_t)p;
      }
      mrow[m] = mnew;
      lrow[m] = lrow[m] * al + ls;
      alphas[m] = al;
    }
    __syncthreads();

    const int moff = (lane < 16) ? 0 : 8;
    float alj[8];
#pragma unroll
    for (int j = 0; j < 8; ++j) alj[j] = alphas[moff + j];
#pragma unroll
    for (int t = 0; t < 4; ++t)
#pragma unroll
      for (int j = 0; j < 8; ++j) acc[t][j] *= alj[j];

    const v16bf pa = wmma_load_a(Pt, 32, lane);
#pragma unroll
    for (int t = 0; t < 4; ++t) {
      // B(k=s, n=d): Vt[(dc0+t*16+n)][jt+k] -> K-contiguous loads
      const v16bf vb = wmma_load_b(Vp + (size_t)(dc0 + t * 16) * kS + jt, kS, lane);
      acc[t] = wmma_bf16(pa, vb, acc[t]);
    }
    __syncthreads();
  }

  // epilogue: out = acc/l + x1 ; xf = DyT_f(out)
  const int n = lane & 15;
  const int moff = (lane < 16) ? 0 : 8;
  const float alF = alphaF[0];
#pragma unroll
  for (int t = 0; t < 4; ++t) {
#pragma unroll
    for (int j = 0; j < 8; ++j) {
      const int row = row0 + moff + j;
      const int col = dc0 + t * 16 + n;
      const size_t g = ((size_t)b * kS + row) * kD + col;
      const float o = acc[t][j] / lrow[moff + j] + x1[g];
      x2[g] = o;
      xfb[g] = (bf16_t)(tanhf(alF * o) * wF[col] + bF[col]);
    }
  }
}

// ---------------------------------------------------------------- kernel 7
// out = xf @ circulant(f_b) + x2   (gelu branch of the ffn is dead code)
__global__ void k_ffn_gemm(const bf16_t* __restrict__ A,
                           const bf16_t* __restrict__ Wf,
                           const float* __restrict__ x2,
                           float* __restrict__ out) {
  const int lane = threadIdx.x & 31;
  const int wave = threadIdx.x >> 5;
  const int row0 = blockIdx.y * 64 + (wave & 3) * 16;
  const int col0 = blockIdx.x * 32 + (wave >> 2) * 16;
  v8f c = {};
  for (int kk = 0; kk < kD; kk += 32) {
    __builtin_prefetch(A + (size_t)row0 * kD + kk + 64, 0, 1);
    const v16bf a = wmma_load_a(A + (size_t)row0 * kD + kk, kD, lane);
    const v16bf bw = wmma_load_b(Wf + (size_t)col0 * kD + kk, kD, lane);
    c = wmma_bf16(a, bw, c);
  }
  const int n = lane & 15;
  const int moff = (lane < 16) ? 0 : 8;
#pragma unroll
  for (int j = 0; j < 8; ++j) {
    const size_t g = (size_t)(row0 + moff + j) * kD + (col0 + n);
    out[g] = c[j] + x2[g];
  }
}

// ---------------------------------------------------------------- launcher
extern "C" void kernel_launch(void* const* d_in, const int* in_sizes, int n_in,
                              void* d_out, int out_size, void* d_ws, size_t ws_size,
                              hipStream_t stream) {
  (void)in_sizes; (void)n_in; (void)out_size; (void)ws_size;
  const float* x      = (const float*)d_in[0];
  const float* a_qr   = (const float*)d_in[1];
  const float* a_kr   = (const float*)d_in[2];
  const float* a_vr   = (const float*)d_in[3];
  const float* a_rtr  = (const float*)d_in[4];
  const float* b_qr   = (const float*)d_in[5];
  const float* b_kr   = (const float*)d_in[6];
  const float* b_vr   = (const float*)d_in[7];
  /* d_in[8] = f_ar : dead code in reference */
  const float* f_br   = (const float*)d_in[9];
  const float* a_qi   = (const float*)d_in[10];
  const float* a_ki   = (const float*)d_in[11];
  const float* a_vi   = (const float*)d_in[12];
  const float* a_rti  = (const float*)d_in[13];
  const float* b_qi   = (const float*)d_in[14];
  const float* b_ki   = (const float*)d_in[15];
  const float* b_vi   = (const float*)d_in[16];
  /* d_in[17] = f_ai : dead code */
  const float* f_bi   = (const float*)d_in[18];
  /* d_in[19] = f_bias : dead code */
  const float* dyta_a = (const float*)d_in[20];
  const float* dyta_w = (const float*)d_in[21];
  const float* dyta_b = (const float*)d_in[22];
  const float* dytb_a = (const float*)d_in[23];
  const float* dytb_w = (const float*)d_in[24];
  const float* dytb_b = (const float*)d_in[25];
  const float* dytf_a = (const float*)d_in[26];
  const float* dytf_w = (const float*)d_in[27];
  const float* dytf_b = (const float*)d_in[28];

  // workspace carve-up (complex buffers are double-used via stream ordering:
  //   bufA: Xf -> Yt ;  bufB: Xt -> Yf)
  char* p = (char*)d_ws;
  auto take = [&](size_t bytes) -> char* {
    char* r = p;
    p += (bytes + 255) & ~(size_t)255;
    return r;
  };
  float2* bufA = (float2*)take(kBSD * sizeof(float2));  // 33.5 MB
  float2* bufB = (float2*)take(kBSD * sizeof(float2));  // 33.5 MB
  float*  x1   = (float*) take(kBSD * sizeof(float));   // 16.8 MB
  float*  x2   = (float*) take(kBSD * sizeof(float));   // 16.8 MB
  bf16_t* xb   = (bf16_t*)take(kBSD * sizeof(bf16_t));  //  8.4 MB
  bf16_t* xf   = (bf16_t*)take(kBSD * sizeof(bf16_t));  //  8.4 MB
  bf16_t* Qb   = (bf16_t*)take(kBSD * sizeof(bf16_t));  //  8.4 MB
  bf16_t* Kb   = (bf16_t*)take(kBSD * sizeof(bf16_t));  //  8.4 MB
  bf16_t* Vt   = (bf16_t*)take(kBSD * sizeof(bf16_t));  //  8.4 MB (transposed)
  bf16_t* Mtq  = (bf16_t*)take((size_t)kD * kD * sizeof(bf16_t));
  bf16_t* Mtk  = (bf16_t*)take((size_t)kD * kD * sizeof(bf16_t));
  bf16_t* Mtv  = (bf16_t*)take((size_t)kD * kD * sizeof(bf16_t));
  bf16_t* Mtf  = (bf16_t*)take((size_t)kD * kD * sizeof(bf16_t));

  float2* Xf = bufA;   // [B][S][D]
  float2* Xt = bufB;   // [B][D][S]
  float2* Yt = bufA;   // [B][D][S]  (reuses Xf after transpose)
  float2* Yf = bufB;   // [B][S][D]  (reuses Xt after kernel 3)
  float* outp = (float*)d_out;

  // 1) circulant matrices for sdpa q/k/v + ffn
  k_build_circulant<<<4, kD, 0, stream>>>(b_qr, b_qi, Mtq,
                                          b_kr, b_ki, Mtk,
                                          b_vr, b_vi, Mtv,
                                          f_br, f_bi, Mtf);
  // 2) DyT_a + FFT along D
  k_dyt_fft_d<<<kBS, 256, 0, stream>>>(x, dyta_a, dyta_w, dyta_b, Xf);
  // 2b) transpose Xf [S][D] -> Xt [D][S] (coalesced S-axis FFT input)
  {
    dim3 grid(kD / 32, kS / 32, kB);
    k_transpose_c<<<grid, 256, 0, stream>>>(Xf, Xt, kS, kD);
  }
  // 3) spectral attention along S (contiguous columns)
  k_attn_a_scol<<<kB * kD, 512, 0, stream>>>(Xt,
                                             a_qr, a_qi, a_kr, a_ki,
                                             a_vr, a_vi, a_rtr, a_rti, Yt);
  // 3b) transpose Yt [D][S] -> Yf [S][D]
  {
    dim3 grid(kS / 32, kD / 32, kB);
    k_transpose_c<<<grid, 256, 0, stream>>>(Yt, Yf, kD, kS);
  }
  // 4) IFFT along D + residual + DyT_b
  k_ifft_d_res<<<kBS, 256, 0, stream>>>(Yf, x, dytb_a, dytb_w, dytb_b, x1, xb);
  // 5) q/k/v circulant projections (bf16 WMMA), V stored transposed
  {
    dim3 grid(kD / 32, kBS / 64);
    k_qkv_gemm<<<grid, 256, 0, stream>>>(xb, Mtq, Mtk, Mtv, Qb, Kb, Vt);
  }
  // 6) flash attention + residual + DyT_f
  {
    dim3 grid(kS / 16, kB);
    k_flash_attn<<<grid, 256, 0, stream>>>(Qb, Kb, Vt, x1,
                                           dytf_a, dytf_w, dytf_b, x2, xf);
  }
  // 7) ffn circulant projection + final residual -> d_out
  {
    dim3 grid(kD / 32, kBS / 64);
    k_ffn_gemm<<<grid, 256, 0, stream>>>(xf, Mtf, x2, outp);
  }
}
